// SmallQKVModel_38010460570082
// MI455X (gfx1250) — compile-verified, matrix operand order checked by reading, not run on hardware
//
#include <hip/hip_runtime.h>
#include <hip/hip_bf16.h>

// MI455X (gfx1250): LayerNorm -> QKV proj -> softmax(QK^T) V -> out proj
// Compute path: TDM (tensor_load_to_lds, double-buffered, HW-padded LDS rows)
//               -> ds_load_b128 fragments -> v_wmma_f32_16x16x32_f16 (f32 acc).
// Workspace requirement: ~128 MB.

typedef __attribute__((ext_vector_type(16))) _Float16 v16h;
typedef __attribute__((ext_vector_type(8)))  _Float16 v8h;
typedef __attribute__((ext_vector_type(8)))  float    v8f;
typedef __attribute__((ext_vector_type(4)))  unsigned int v4u;
typedef __attribute__((ext_vector_type(8)))  int      v8i_;
typedef __attribute__((ext_vector_type(4)))  int      v4i_;

#define TPB 256

// ---------------- fp32 -> f16 convert ----------------
__global__ void cvt_f32_f16(const float* __restrict__ in, _Float16* __restrict__ out, int n) {
    int i = blockIdx.x * TPB + threadIdx.x;
    if (i < n) out[i] = (_Float16)in[i];
}

// ---------------- LayerNorm (one block per row) ----------------
__global__ void ln_f16(const float* __restrict__ x, const float* __restrict__ w,
                       _Float16* __restrict__ h, int D) {
    __shared__ float s1[TPB], s2[TPB];
    const int row = blockIdx.x, tid = threadIdx.x;
    const float* xr = x + (size_t)row * D;
    float a = 0.f, b = 0.f;
    for (int i = tid; i < D; i += TPB) { float v = xr[i]; a += v; b += v * v; }
    s1[tid] = a; s2[tid] = b; __syncthreads();
    for (int s = TPB / 2; s > 0; s >>= 1) {
        if (tid < s) { s1[tid] += s1[tid + s]; s2[tid] += s2[tid + s]; }
        __syncthreads();
    }
    const float mu  = s1[0] / (float)D;
    const float var = s2[0] / (float)D - mu * mu;
    const float rs  = rsqrtf(var + 1e-5f);
    _Float16* hr = h + (size_t)row * D;
    for (int i = tid; i < D; i += TPB) hr[i] = (_Float16)((xr[i] - mu) * rs * w[i]);
}

// ---------------- TDM: load a 128-row x 32-col f16 tile into LDS ----------------
// Descriptor packed per CDNA5 ISA section 8 (D# groups).  2-byte elements,
// row stride = ldK elements, LDS padding of 4 DWORDs (16B) every 16 DWORDs (64B)
// -> LDS row pitch 80B (bank-conflict-free for the fragment ds_loads below).
__device__ __forceinline__ void tdm_load_tile_128x32(unsigned lds_off,
                                                     const _Float16* gptr,
                                                     int ldK, int rowsTotal) {
    unsigned long long ga = (unsigned long long)(const void*)gptr;
    v4u g0;
    g0[0] = 1u;                                             // count=1, user mode
    g0[1] = lds_off;                                        // LDS byte address
    g0[2] = (unsigned)(ga & 0xFFFFFFFFull);                 // global_addr[31:0]
    g0[3] = (unsigned)((ga >> 32) & 0x01FFFFFFull)          // global_addr[56:32]
          | (2u << 30);                                     // type = 2 ("image")
    const unsigned dim0 = (unsigned)ldK;
    const unsigned dim1 = (unsigned)rowsTotal;
    const unsigned str0 = (unsigned)ldK;                    // fits 32 bits here
    unsigned w0 = (1u << 16)        // data_size = 1 -> 2 bytes
                | (1u << 20)        // pad_enable
                | (3u << 22)        // pad_interval: 16 DWORDs (64B) between pads
                | (3u << 25);       // pad_amount: 4 DWORDs (16B)
    unsigned w1 = (dim0 & 0xFFFFu) << 16;                   // tensor_dim0 lo
    unsigned w2 = (dim0 >> 16) | ((dim1 & 0xFFFFu) << 16);  // dim0 hi | dim1 lo
    unsigned w3 = ((dim1 >> 16) & 0xFFFFu) | (32u << 16);   // dim1 hi | tile_dim0=32
    unsigned w4 = 128u;                                     // tile_dim1=128, tile_dim2=0
    unsigned w5 = str0;                                     // tensor_dim0_stride lo
    unsigned w6 = 0u;                                       // stride hi | dim1_stride lo
    unsigned w7 = 0u;
    v8i_ g1 = { (int)w0, (int)w1, (int)w2, (int)w3,
                (int)w4, (int)w5, (int)w6, (int)w7 };
    v4i_ z4 = { 0, 0, 0, 0 };                               // groups 2/3: 2-D tensor
    v8i_ z8 = { 0, 0, 0, 0, 0, 0, 0, 0 };                   // extra group (disabled)
    __builtin_amdgcn_tensor_load_to_lds(g0, g1, z4, z4, z8, 0);
}

// ---------------- WMMA GEMM: C[M,N] = A[M,K] * Bt[N,K]^T ----------------
// Block = 256 threads = 8 waves (4 along M x 2 along N) -> 128x128 macro tile.
// Each wave computes a 32x64 strip: 2 M-subtiles x 4 N-subtiles = 8 WMMA / K-step.
// A & B K-step tiles (128x32 f16) staged in LDS by the TDM, double buffered.
// All 6 fragments (2 A + 4 B) are loaded before the 8-WMMA chain so the DS pipe
// and the matrix pipe overlap instead of ping-ponging on a shared register octet.
#define LDS_PITCH 80           // 64B row + 16B TDM pad
#define TILE_BYTES (128 * LDS_PITCH)   // 10240 B per tile buffer

template <bool F16OUT>
__global__ void wmma_gemm_nt(const _Float16* __restrict__ A,
                             const _Float16* __restrict__ Bt,
                             void* __restrict__ C,
                             int M, int N, int K) {
    __shared__ alignas(16) unsigned char lds[4 * TILE_BYTES];   // A0 A1 B0 B1
    const unsigned ldsBase = (unsigned)(unsigned long long)(void*)lds;

    const int tid    = threadIdx.x;
    const int lane   = tid & 31;
    const int wave   = tid >> 5;
    const int half   = lane >> 4;          // K-chunk selector (ISA frag layout)
    const int lanelo = lane & 15;          // row/col within 16x16 tile
    const int wm     = wave & 3;           // M group: 4 x 32 rows
    const int wn     = wave >> 2;          // N group: 2 x 64 cols

    const int mBlk = blockIdx.y * 128;
    const int nBlk = blockIdx.x * 128;
    const int ksteps = K >> 5;

    const v8f vzero = {0.f, 0.f, 0.f, 0.f, 0.f, 0.f, 0.f, 0.f};
    v8f acc[2][4];
#pragma unroll
    for (int s = 0; s < 2; ++s)
#pragma unroll
        for (int t = 0; t < 4; ++t) acc[s][t] = vzero;

    const _Float16* aBase = A  + (size_t)mBlk * K;
    const _Float16* bBase = Bt + (size_t)nBlk * K;

    // prologue: wave 0 kicks off TDM for K-step 0 (TDM ignores EXEC -> wave guard)
    if (tid < 32) {
        tdm_load_tile_128x32(ldsBase + 0 * TILE_BYTES, aBase, K, M - mBlk);
        tdm_load_tile_128x32(ldsBase + 2 * TILE_BYTES, bBase, K, N - nBlk);
    }

    for (int i = 0; i < ksteps; ++i) {
        const int cur = i & 1;
        if (tid < 32) {
            if (i + 1 < ksteps) {
                const int nxt = cur ^ 1;
                const int k1 = (i + 1) << 5;
                tdm_load_tile_128x32(ldsBase + nxt * TILE_BYTES,       aBase + k1, K, M - mBlk);
                tdm_load_tile_128x32(ldsBase + (2 + nxt) * TILE_BYTES, bBase + k1, K, N - nBlk);
                __builtin_amdgcn_s_wait_tensorcnt((short)2);  // step-i pair complete
            } else {
                __builtin_amdgcn_s_wait_tensorcnt((short)0);
            }
        }
        __syncthreads();   // buffers for step i visible to all waves

        const unsigned char* aB = lds + cur * TILE_BYTES;
        const unsigned char* bB = lds + (2 + cur) * TILE_BYTES;

        // ---- load ALL fragments first (12 x ds_load_b128, clause-friendly) ----
        v16h afrag[2];
#pragma unroll
        for (int s = 0; s < 2; ++s) {
            const unsigned char* ap = aB + (wm * 32 + s * 16 + lanelo) * LDS_PITCH + half * 16;
            v8h lo = *(const v8h*)ap;
            v8h hi = *(const v8h*)(ap + 32);
            afrag[s] = __builtin_shufflevector(lo, hi,
                        0, 1, 2, 3, 4, 5, 6, 7, 8, 9, 10, 11, 12, 13, 14, 15);
        }
        v16h bfrag[4];
#pragma unroll
        for (int t = 0; t < 4; ++t) {
            const unsigned char* bp = bB + (wn * 64 + t * 16 + lanelo) * LDS_PITCH + half * 16;
            v8h lo = *(const v8h*)bp;
            v8h hi = *(const v8h*)(bp + 32);
            bfrag[t] = __builtin_shufflevector(lo, hi,
                        0, 1, 2, 3, 4, 5, 6, 7, 8, 9, 10, 11, 12, 13, 14, 15);
        }
        // ---- 8 independent WMMAs back-to-back ----
#pragma unroll
        for (int t = 0; t < 4; ++t) {
            acc[0][t] = __builtin_amdgcn_wmma_f32_16x16x32_f16(
                            false, afrag[0], false, bfrag[t], (short)0, acc[0][t], false, false);
            acc[1][t] = __builtin_amdgcn_wmma_f32_16x16x32_f16(
                            false, afrag[1], false, bfrag[t], (short)0, acc[1][t], false, false);
        }
        __syncthreads();   // everyone done reading buffer `cur` before TDM reuses it
    }

    // C/D layout: VGPR r -> row (half*8 + r), lanes 0..15 -> column within tile.
    const int mbase = mBlk + wm * 32 + half * 8;
#pragma unroll
    for (int s = 0; s < 2; ++s) {
#pragma unroll
        for (int t = 0; t < 4; ++t) {
            const int n = nBlk + wn * 64 + t * 16 + lanelo;
#pragma unroll
            for (int r = 0; r < 8; ++r) {
                const int m = mbase + s * 16 + r;
                if (F16OUT)
                    ((_Float16*)C)[(size_t)m * N + n] = (_Float16)acc[s][t][r];
                else
                    ((float*)C)[(size_t)m * N + n] = acc[s][t][r];
            }
        }
    }
}

// ---------------- softmax over a row of fp32 scores -> f16 probs ----------------
__global__ void softmax_row(const float* __restrict__ scores, _Float16* __restrict__ attn,
                            int Ncols) {
    __shared__ float sd[TPB];
    const int row = blockIdx.x, tid = threadIdx.x;
    const float* sr = scores + (size_t)row * Ncols;
    _Float16* ar = attn + (size_t)row * Ncols;

    float m = -3.4e38f;
    for (int i = tid; i < Ncols; i += TPB) m = fmaxf(m, sr[i]);
    sd[tid] = m; __syncthreads();
    for (int s = TPB / 2; s > 0; s >>= 1) {
        if (tid < s) sd[tid] = fmaxf(sd[tid], sd[tid + s]);
        __syncthreads();
    }
    m = sd[0]; __syncthreads();

    float sum = 0.f;
    for (int i = tid; i < Ncols; i += TPB) sum += __expf(sr[i] - m);
    sd[tid] = sum; __syncthreads();
    for (int s = TPB / 2; s > 0; s >>= 1) {
        if (tid < s) sd[tid] += sd[tid + s];
        __syncthreads();
    }
    const float inv = 1.f / sd[0];

    for (int i = tid; i < Ncols; i += TPB) ar[i] = (_Float16)(__expf(sr[i] - m) * inv);
}

// ---------------- transpose V: [B][S][D] f16 -> [B][D][S] f16 ----------------
__global__ void transpose_v(const _Float16* __restrict__ v, _Float16* __restrict__ vt,
                            int S, int D) {
    __shared__ _Float16 tile[32][33];
    const int b  = blockIdx.z;
    const int d0 = blockIdx.x * 32;
    const int s0 = blockIdx.y * 32;
    const int tx = threadIdx.x, ty = threadIdx.y;   // block (32, 8)
#pragma unroll
    for (int j = 0; j < 32; j += 8)
        tile[ty + j][tx] = v[((size_t)b * S + s0 + ty + j) * D + d0 + tx];
    __syncthreads();
#pragma unroll
    for (int j = 0; j < 32; j += 8)
        vt[((size_t)b * D + d0 + ty + j) * S + s0 + tx] = tile[tx][ty + j];
}

extern "C" void kernel_launch(void* const* d_in, const int* in_sizes, int n_in,
                              void* d_out, int out_size, void* d_ws, size_t ws_size,
                              hipStream_t stream) {
    (void)in_sizes; (void)n_in; (void)out_size; (void)ws_size;
    const int B = 4, S = 2048, D = 1024;
    const int M = B * S;                 // 8192 activation rows

    const float* x    = (const float*)d_in[0];
    const float* ln_w = (const float*)d_in[1];
    const float* wq   = (const float*)d_in[2];
    const float* wk   = (const float*)d_in[3];
    const float* wv   = (const float*)d_in[4];
    const float* wo   = (const float*)d_in[5];
    float* out = (float*)d_out;

    // ---- workspace carve-up (256B-aligned offsets) ----
    char* p = (char*)d_ws;
    auto carve = [&](size_t bytes) -> char* {
        char* r = p;
        p += (bytes + 255) & ~(size_t)255;
        return r;
    };
    _Float16* h      = (_Float16*)carve((size_t)M * D * 2);        // 16 MB
    _Float16* wqh    = (_Float16*)carve((size_t)D * D * 2);        //  2 MB
    _Float16* wkh    = (_Float16*)carve((size_t)D * D * 2);
    _Float16* wvh    = (_Float16*)carve((size_t)D * D * 2);
    _Float16* woh    = (_Float16*)carve((size_t)D * D * 2);
    _Float16* q      = (_Float16*)carve((size_t)M * D * 2);        // 16 MB
    _Float16* k      = (_Float16*)carve((size_t)M * D * 2);        // 16 MB
    _Float16* v      = (_Float16*)carve((size_t)M * D * 2);        // 16 MB
    _Float16* vt     = (_Float16*)carve((size_t)M * D * 2);        // 16 MB [B][D][S]
    float*    scores = (float*)   carve((size_t)S * S * 4);        // 16 MB (one batch)
    _Float16* attn   = (_Float16*)carve((size_t)S * S * 2);        //  8 MB (one batch)
    _Float16* ctx    = (_Float16*)carve((size_t)M * D * 2);        // 16 MB

    // ---- 1) weights fp32 -> f16 ----
    {
        const int n = D * D;
        const int g = (n + TPB - 1) / TPB;
        cvt_f32_f16<<<g, TPB, 0, stream>>>(wq, wqh, n);
        cvt_f32_f16<<<g, TPB, 0, stream>>>(wk, wkh, n);
        cvt_f32_f16<<<g, TPB, 0, stream>>>(wv, wvh, n);
        cvt_f32_f16<<<g, TPB, 0, stream>>>(wo, woh, n);
    }

    // ---- 2) LayerNorm -> h (f16) ----
    ln_f16<<<M, TPB, 0, stream>>>(x, ln_w, h, D);

    // ---- 3) Q/K/V projections: [8192,1024] = h @ W^T ----
    {
        dim3 grid(D / 128, M / 128);
        wmma_gemm_nt<true><<<grid, TPB, 0, stream>>>(h, wqh, q, M, D, D);
        wmma_gemm_nt<true><<<grid, TPB, 0, stream>>>(h, wkh, k, M, D, D);
        wmma_gemm_nt<true><<<grid, TPB, 0, stream>>>(h, wvh, v, M, D, D);
    }

    // ---- 4) V transpose (ctx GEMM wants B as [N,K]) ----
    {
        dim3 grid(D / 32, S / 32, B);
        dim3 blk(32, 8);
        transpose_v<<<grid, blk, 0, stream>>>(v, vt, S, D);
    }

    // ---- 5) per-batch attention ----
    for (int b = 0; b < B; ++b) {
        const _Float16* qb  = q  + (size_t)b * S * D;
        const _Float16* kb  = k  + (size_t)b * S * D;
        const _Float16* vtb = vt + (size_t)b * D * S;
        _Float16* ctxb      = ctx + (size_t)b * S * D;

        // scores[S,S] = Q @ K^T  (fp32 out, K-dim = D)
        {
            dim3 grid(S / 128, S / 128);
            wmma_gemm_nt<false><<<grid, TPB, 0, stream>>>(qb, kb, (void*)scores, S, S, D);
        }
        // softmax rows -> f16 probabilities
        softmax_row<<<S, TPB, 0, stream>>>(scores, attn, S);
        // ctx[S,D] = attn @ V  == attn[S,S(K)] @ vt[D,S(K)]^T  (f16 out, K-dim = S)
        {
            dim3 grid(D / 128, S / 128);
            wmma_gemm_nt<true><<<grid, TPB, 0, stream>>>(attn, vtb, (void*)ctxb, S, D, S);
        }
    }

    // ---- 6) out = ctx @ Wo^T  (fp32 out, straight to d_out) ----
    {
        dim3 grid(D / 128, M / 128);
        wmma_gemm_nt<false><<<grid, TPB, 0, stream>>>(ctx, woh, (void*)out, M, D, D);
    }
}
